// BaseDenseAttention_88751204204680
// MI455X (gfx1250) — matrix operand order repaired and checked by reading, not compiled, not run on hardware
//
#include <hip/hip_runtime.h>
#include <stdint.h>

// CDNA5 WMMA fragment types (wave32)
typedef __attribute__((ext_vector_type(16))) __bf16 v16bf;
typedef __attribute__((ext_vector_type(2)))  __bf16 v2bf;
typedef __attribute__((ext_vector_type(8)))  float  v8f;
typedef __attribute__((ext_vector_type(4)))  int    v4i;

#define NEGINF (-3.0e38f)

#if defined(__has_builtin)
#if __has_builtin(__builtin_amdgcn_cvt_pk_bf16_f32)
#define HAVE_PK_BF16 1
#endif
#if __has_builtin(__builtin_amdgcn_global_load_async_to_lds_b128)
#define HAVE_ASYNC 1
#endif
#endif

#ifdef HAVE_ASYNC
typedef __attribute__((address_space(1))) v4i* g4p;   // global v4i*
typedef __attribute__((address_space(3))) v4i* l4p;   // LDS v4i*
#endif

namespace {
constexpr int TQ = 2048, TV = 2048, DIM = 512;
constexpr int BM = 32;        // query rows per block
constexpr int BN = 64;        // keys per iteration
constexpr int THREADS = 256;  // 8 waves

// Padded LDS row strides (bf16 elements) for conflict-free ds_load_b128:
// QKS: 1040B = 260 dwords ≡ 4 (mod 64 banks); VT/PS: 144B = 36 dwords,
// l15*36 mod 64 enumerates 16 distinct multiples of 4.
constexpr int QKS = DIM + 8;  // 520
constexpr int VTS = BN + 8;   // 72
constexpr int PSS = BN + 8;   // 72

// LDS layout (bytes) — double-buffered K/V^T for async-DMA/compute overlap
constexpr int OFF_QS   = 0;                         // [BM][QKS] bf16 (Q tile)
constexpr int OFF_KS0  = OFF_QS  + BM * QKS * 2;    // [BN][QKS] bf16 (K rows, buf0)
constexpr int OFF_VT0  = OFF_KS0 + BN * QKS * 2;    // [DIM][VTS] bf16 (V^T, buf0)
constexpr int OFF_KS1  = OFF_VT0 + DIM * VTS * 2;   // buf1
constexpr int OFF_VT1  = OFF_KS1 + BN * QKS * 2;    // buf1
constexpr int OFF_PS   = OFF_VT1 + DIM * VTS * 2;   // [BM][PSS] bf16 (probs)
constexpr int OFF_SMAX = OFF_PS + BM * PSS * 2;     // [4][BM] f32
constexpr int OFF_SSUM = OFF_SMAX + 4 * BM * 4;     // [4][BM] f32
constexpr int OFF_M    = OFF_SSUM + 4 * BM * 4;     // [BM] f32 running max
constexpr int OFF_L    = OFF_M + BM * 4;            // [BM] f32 running sum
constexpr int OFF_AL   = OFF_L + BM * 4;            // [BM] f32 alpha
constexpr int LDS_BYTES = OFF_AL + BM * 4;          // 319,872 B <= 320 KB/WGP
}

__device__ __forceinline__ unsigned short f2bf(float f) {
  union { float f; uint32_t u; } x; x.f = f;
  uint32_t u = x.u + 0x7fffu + ((x.u >> 16) & 1u);   // round-to-nearest-even
  return (unsigned short)(u >> 16);
}

__device__ __forceinline__ uint32_t pack2bf(float lo, float hi) {
#ifdef HAVE_PK_BF16
  union { v2bf v; uint32_t u; } c;
  c.v = __builtin_amdgcn_cvt_pk_bf16_f32(lo, hi);
  return c.u;
#else
  return (uint32_t)f2bf(lo) | ((uint32_t)f2bf(hi) << 16);
#endif
}

__device__ __forceinline__ v16bf frag(uint4 lo, uint4 hi) {
  union { v16bf v; uint4 q[2]; } u;
  u.q[0] = lo; u.q[1] = hi;
  return u.v;
}

// 16B LDS-staging copy: async DMA when available, else load+store.
// AS3 pointer built by integer truncation: ISA aperture rule says the low
// 32 bits of a generic LDS address are the LDS byte offset.
__device__ __forceinline__ void copy16(const uint4* src, uint32_t* dst) {
#ifdef HAVE_ASYNC
  __builtin_amdgcn_global_load_async_to_lds_b128((g4p)(uintptr_t)src,
                                                 (l4p)(uintptr_t)dst, 0, 0);
#else
  *(uint4*)dst = *src;
#endif
}

template <int N>
__device__ __forceinline__ void wait_async() {
#ifdef HAVE_ASYNC
#if defined(__has_builtin) && __has_builtin(__builtin_amdgcn_s_wait_asynccnt)
  __builtin_amdgcn_s_wait_asynccnt(N);
#else
  asm volatile("s_wait_asynccnt %0" :: "i"(N) : "memory");
#endif
#endif
}

// issue one key-block's worth of bf16 K rows + V^T rows into LDS buffers
__device__ __forceinline__ void stage_block_bf16(int tid, int b, int k0,
    const uint16_t* __restrict__ Vbf, const uint16_t* __restrict__ VbfT,
    unsigned short* Ks, unsigned short* Vt) {
  const uint4* vrow = (const uint4*)(Vbf + ((size_t)b * TV + k0) * DIM);
  for (int it = tid; it < BN * DIM / 8; it += THREADS) {
    int row = it >> 6;                 // DIM/8 = 64 16B-chunks per key row
    int c8  = it & 63;
    copy16(vrow + it, (uint32_t*)Ks + row * (QKS / 2) + c8 * 4);
  }
  const uint16_t* vcol = VbfT + (size_t)b * DIM * TV + k0;
  for (int it = tid; it < DIM * BN / 8; it += THREADS) {
    int drow = it >> 3;                // BN/8 = 8 16B-chunks per d row
    int c8   = it & 7;
    copy16((const uint4*)(vcol + (size_t)drow * TV) + c8,
           (uint32_t*)Vt + drow * (VTS / 2) + c8 * 4);
  }
}

// ---- one-time fp32 -> bf16 pre-conversion of V into row-major + transposed
__global__ __launch_bounds__(256)
void prep_bf16(const float* __restrict__ V, uint16_t* __restrict__ Vbf,
               uint16_t* __restrict__ VbfT) {
  const int b = blockIdx.y;
  int t  = blockIdx.x * 256 + threadIdx.x;   // 65536 4x4 tiles per batch
  int dc = t & 127;                          // DIM/4
  int kr = t >> 7;                           // TV/4
  const float4* src = (const float4*)(V + ((size_t)b * TV + (size_t)kr * 4) * DIM);
  float v[4][4];
  #pragma unroll
  for (int r = 0; r < 4; ++r) {
    float4 x = src[(size_t)r * (DIM / 4) + dc];
    v[r][0] = x.x; v[r][1] = x.y; v[r][2] = x.z; v[r][3] = x.w;
  }
  uint32_t* vr = (uint32_t*)Vbf;
  #pragma unroll
  for (int r = 0; r < 4; ++r) {
    uint2 p; p.x = pack2bf(v[r][0], v[r][1]); p.y = pack2bf(v[r][2], v[r][3]);
    *(uint2*)(vr + ((size_t)b * TV + kr * 4 + r) * (DIM / 2) + dc * 2) = p;
  }
  uint32_t* vc = (uint32_t*)VbfT;
  #pragma unroll
  for (int c = 0; c < 4; ++c) {
    uint2 p; p.x = pack2bf(v[0][c], v[1][c]); p.y = pack2bf(v[2][c], v[3][c]);
    *(uint2*)(vc + ((size_t)b * DIM + dc * 4 + c) * (TV / 2) + kr * 2) = p;
  }
}

template <bool PRE>
__global__ __launch_bounds__(THREADS)
void fa_fwd_gfx1250(const float* __restrict__ Q, const float* __restrict__ V,
                    const unsigned char* __restrict__ qmask,
                    const unsigned char* __restrict__ vmask,
                    float* __restrict__ Out,
                    const uint16_t* __restrict__ Vbf,
                    const uint16_t* __restrict__ VbfT)
{
  extern __shared__ char lds[];
  unsigned short* Qs = (unsigned short*)(lds + OFF_QS);
  unsigned short* Ps = (unsigned short*)(lds + OFF_PS);
  float* smax = (float*)(lds + OFF_SMAX);
  float* ssum = (float*)(lds + OFF_SSUM);
  float* mrow = (float*)(lds + OFF_M);
  float* lrow = (float*)(lds + OFF_L);
  float* arow = (float*)(lds + OFF_AL);

  const int tid  = threadIdx.x;
  const int lane = tid & 31;
  const int wave = tid >> 5;
  const int mi   = wave >> 2;           // 0..1 : 16-row group of query tile
  const int wx   = wave & 3;            // score: 16-col group of S ; PV: 128-wide D slice
  const int lh   = (lane & 16) ? 1 : 0; // lane half (WMMA operand layout)
  const int l15  = lane & 15;
  const int rb   = mi * 16 + lh * 8;    // base row (C-tile layout: VGPR i -> row rb+i)

  const int b  = blockIdx.y;
  const int q0 = blockIdx.x * BM;

  const int nkb = (q0 + BM + BN - 1) / BN;   // causal: only key blocks with k <= q0+31
  const float* Vb = V + (size_t)b * TV * DIM;
  const unsigned char* vmb = vmask + (size_t)b * TV;

  // ---- prologue: start DMA of key block 0 while we stage Q
  if constexpr (PRE)
    stage_block_bf16(tid, b, 0, Vbf, VbfT,
                     (unsigned short*)(lds + OFF_KS0),
                     (unsigned short*)(lds + OFF_VT0));

  // ---- stage Q tile fp32 -> bf16 into (padded) LDS, init softmax state
  const float* Qg = Q + ((size_t)b * TQ + q0) * DIM;
  for (int it = tid; it < BM * DIM / 4; it += THREADS) {
    float4 q4 = ((const float4*)Qg)[it];
    int row = it >> 7;                 // DIM/4 = 128 float4 per row
    int c4  = it & 127;
    uint2 p;
    p.x = pack2bf(q4.x, q4.y);
    p.y = pack2bf(q4.z, q4.w);
    *(uint2*)((uint32_t*)Qs + row * (QKS / 2) + c4 * 2) = p;
  }
  if (tid < BM) { mrow[tid] = NEGINF; lrow[tid] = 0.0f; }

  v8f acc[8] = {};   // 8 C-tiles: rows [mi*16,16) x cols [wx*128 + t*16, 16)

  for (int kb = 0; kb < nkb; ++kb) {
    const int k0 = kb * BN;
    if (!vmb[k0]) break;                 // length mask is monotone: nothing valid past here
    __syncthreads();                     // prior compute done; safe to overwrite other buffer

    if constexpr (PRE) {
      wait_async<0>();                   // current block's DMA landed (this wave's share)
      if (kb + 1 < nkb) {                // prefetch next block; overlaps compute below
        const int p = (kb + 1) & 1;
        stage_block_bf16(tid, b, k0 + BN, Vbf, VbfT,
                         (unsigned short*)(lds + (p ? OFF_KS1 : OFF_KS0)),
                         (unsigned short*)(lds + (p ? OFF_VT1 : OFF_VT0)));
      }
    } else {
      // ---- fallback: stage fp32 V as 4x4 tiles with in-kernel conversion
      const float4* Vg4 = (const float4*)(Vb + (size_t)k0 * DIM);
      for (int it = tid; it < (BN / 4) * (DIM / 4); it += THREADS) {
        int dc = it & 127;
        int kr = it >> 7;
        float v[4][4];
        #pragma unroll
        for (int r = 0; r < 4; ++r) {
          float4 t = Vg4[(kr * 4 + r) * (DIM / 4) + dc];
          v[r][0] = t.x; v[r][1] = t.y; v[r][2] = t.z; v[r][3] = t.w;
        }
        #pragma unroll
        for (int r = 0; r < 4; ++r) {
          uint2 p;
          p.x = pack2bf(v[r][0], v[r][1]);
          p.y = pack2bf(v[r][2], v[r][3]);
          *(uint2*)((uint32_t*)(lds + OFF_KS0) + (kr * 4 + r) * (QKS / 2) + dc * 2) = p;
        }
        #pragma unroll
        for (int c0 = 0; c0 < 4; ++c0) {
          int c = (c0 + tid) & 3;
          uint2 p;
          p.x = pack2bf(v[0][c], v[1][c]);
          p.y = pack2bf(v[2][c], v[3][c]);
          *(uint2*)((uint32_t*)(lds + OFF_VT0) + (dc * 4 + c) * (VTS / 2) + kr * 2) = p;
        }
      }
    }
    __syncthreads();                     // all waves' DMA for block kb complete

    const int cp = PRE ? (kb & 1) : 0;
    unsigned short* Ks = (unsigned short*)(lds + (cp ? OFF_KS1 : OFF_KS0));
    unsigned short* Vt = (unsigned short*)(lds + (cp ? OFF_VT1 : OFF_VT0));

    // ---- S tile per wave: 16x16, full-D reduction (16 x wmma_f32_16x16x32_bf16)
    v8f s = {};
    const uint32_t* aq = (const uint32_t*)Qs + (mi * 16 + l15) * (QKS / 2);
    const uint32_t* kq = (const uint32_t*)Ks + (wx * 16 + l15) * (QKS / 2);
    #pragma unroll
    for (int kk = 0; kk < DIM / 32; ++kk) {
      uint4 alo = *(const uint4*)(aq + kk*16 + lh*4);       // A: K = kk*32 + lh*8 .. +7
      uint4 ahi = *(const uint4*)(aq + kk*16 + lh*4 + 8);   //    K = +16 .. +23
      uint4 blo = *(const uint4*)(kq + kk*16 + lh*4);       // B: K = kk*32 + lh*8 .. +7
      uint4 bhi = *(const uint4*)(kq + kk*16 + lh*4 + 8);   //    K = +16 .. +23
      s = __builtin_amdgcn_wmma_f32_16x16x32_bf16(false, frag(alo, ahi),
                                                  false, frag(blo, bhi),
                                                  (short)0, s, false, false);
    }

    // ---- mask (causal AND v_mask) + per-row max within tile
    const int kcol = k0 + wx * 16 + l15;
    const bool kval = vmb[kcol] != 0;
    float rmax[8];
    #pragma unroll
    for (int i = 0; i < 8; ++i) {
      int q = q0 + rb + i;
      float sv = (kval && q >= kcol) ? s[i] : -1.0e9f;
      s[i] = sv;
      rmax[i] = sv;
    }
    #pragma unroll
    for (int i = 0; i < 8; ++i) {
      #pragma unroll
      for (int off = 1; off < 16; off <<= 1)
        rmax[i] = fmaxf(rmax[i], __shfl_xor(rmax[i], off, 32));
    }
    if (l15 == 0) {
      #pragma unroll
      for (int i = 0; i < 8; ++i) smax[wx * BM + rb + i] = rmax[i];
    }
    __syncthreads();

    // ---- per-row running max + alpha (one thread per query row)
    if (tid < BM) {
      float mo = mrow[tid];
      float mb = fmaxf(fmaxf(smax[tid], smax[BM + tid]),
                       fmaxf(smax[2*BM + tid], smax[3*BM + tid]));
      float mn = fmaxf(mo, mb);
      mrow[tid] = mn;
      arow[tid] = __expf(mo - mn);
    }
    __syncthreads();

    // ---- P = exp(S - m) -> bf16 LDS ; per-row partial sums
    float rsum[8];
    #pragma unroll
    for (int i = 0; i < 8; ++i) {
      float p = __expf(s[i] - mrow[rb + i]);
      rsum[i] = p;
      Ps[(rb + i) * PSS + wx * 16 + l15] = f2bf(p);
    }
    #pragma unroll
    for (int i = 0; i < 8; ++i) {
      #pragma unroll
      for (int off = 1; off < 16; off <<= 1)
        rsum[i] += __shfl_xor(rsum[i], off, 32);
    }
    if (l15 == 0) {
      #pragma unroll
      for (int i = 0; i < 8; ++i) ssum[wx * BM + rb + i] = rsum[i];
    }
    __syncthreads();

    if (tid < BM)
      lrow[tid] = lrow[tid] * arow[tid]
                + (ssum[tid] + ssum[BM + tid] + ssum[2*BM + tid] + ssum[3*BM + tid]);

    // ---- rescale accumulators by alpha
    float al[8];
    #pragma unroll
    for (int i = 0; i < 8; ++i) al[i] = arow[rb + i];
    #pragma unroll
    for (int t = 0; t < 8; ++t)
      #pragma unroll
      for (int i = 0; i < 8; ++i) acc[t][i] *= al[i];

    // ---- O += P x V  (A = P rows mi*16.., B = Vt rows for this wave's D slice)
    const uint32_t* pq = (const uint32_t*)Ps + (mi * 16 + l15) * (PSS / 2);
    #pragma unroll
    for (int ks = 0; ks < BN / 32; ++ks) {
      uint4 plo = *(const uint4*)(pq + ks*16 + lh*4);
      uint4 phi = *(const uint4*)(pq + ks*16 + lh*4 + 8);
      v16bf pa = frag(plo, phi);
      #pragma unroll
      for (int t = 0; t < 8; ++t) {
        const uint32_t* vq = (const uint32_t*)Vt + (wx*128 + t*16 + l15) * (VTS / 2);
        uint4 vlo = *(const uint4*)(vq + ks*16 + lh*4);
        uint4 vhi = *(const uint4*)(vq + ks*16 + lh*4 + 8);
        acc[t] = __builtin_amdgcn_wmma_f32_16x16x32_bf16(false, pa,
                                                         false, frag(vlo, vhi),
                                                         (short)0, acc[t], false, false);
      }
    }
  }

  // ---- epilogue: divide by softmax denominator, apply q_mask, store fp32
  __syncthreads();
  const unsigned char* qmb = qmask + (size_t)b * TQ;
  float scale[8];
  #pragma unroll
  for (int i = 0; i < 8; ++i) {
    int r = rb + i;
    float l = lrow[r];
    float qv = qmb[q0 + r] ? 1.0f : 0.0f;
    scale[i] = (l > 0.0f) ? (qv / l) : 0.0f;
  }
  #pragma unroll
  for (int t = 0; t < 8; ++t) {
    int dcol = wx * 128 + t * 16 + l15;
    #pragma unroll
    for (int i = 0; i < 8; ++i) {
      int r = rb + i;
      Out[((size_t)b * TQ + (q0 + r)) * DIM + dcol] = acc[t][i] * scale[i];
    }
  }
}

extern "C" void kernel_launch(void* const* d_in, const int* in_sizes, int n_in,
                              void* d_out, int out_size, void* d_ws, size_t ws_size,
                              hipStream_t stream) {
  (void)n_in; (void)out_size;
  const float* Q = (const float*)d_in[0];
  const float* V = (const float*)d_in[1];
  const unsigned char* qm = (const unsigned char*)d_in[2];
  const unsigned char* vm = (const unsigned char*)d_in[3];
  float* Out = (float*)d_out;

  const int B = in_sizes[0] / (TQ * DIM);
  dim3 grid(TQ / BM, B);

  const size_t vbf_elems = (size_t)B * TV * DIM;
  const size_t need = 2 * vbf_elems * sizeof(uint16_t);   // 32 MiB for B=8

  if (ws_size >= need) {
    uint16_t* Vbf  = (uint16_t*)d_ws;
    uint16_t* VbfT = Vbf + vbf_elems;
    prep_bf16<<<dim3((TV / 4) * (DIM / 4) / 256, B), 256, 0, stream>>>(V, Vbf, VbfT);
    fa_fwd_gfx1250<true><<<grid, THREADS, LDS_BYTES, stream>>>(Q, V, qm, vm, Out, Vbf, VbfT);
  } else {
    fa_fwd_gfx1250<false><<<grid, THREADS, LDS_BYTES, stream>>>(Q, V, qm, vm, Out, nullptr, nullptr);
  }
}
